// MaskedAttention_30202210026142
// MI455X (gfx1250) — matrix operand order repaired and checked by reading, not compiled, not run on hardware
//
#include <hip/hip_runtime.h>
#include <hip/hip_bf16.h>

typedef __bf16 bf16_t;
typedef __attribute__((ext_vector_type(16))) __bf16 v16bf;
typedef __attribute__((ext_vector_type(8)))  __bf16 v8bf;
typedef __attribute__((ext_vector_type(8)))  float  v8f;

__device__ __forceinline__ v8f wmma_bf16(v16bf a, v16bf b, v8f c) {
  // D = A(16x32 bf16) * B(32x16 bf16) + C(16x16 f32)
  return __builtin_amdgcn_wmma_f32_16x16x32_bf16(false, a, false, b, (short)0, c, false, false);
}

// load 2 x 8 contiguous f32, convert to bf16 fragment halves
__device__ __forceinline__ v16bf load_cvt16_2(const float* __restrict__ p0,
                                              const float* __restrict__ p1) {
  v16bf r;
  #pragma unroll
  for (int i = 0; i < 8; i++) { r[i] = (__bf16)p0[i]; r[i + 8] = (__bf16)p1[i]; }
  return r;
}
__device__ __forceinline__ v16bf load_cvt16(const float* __restrict__ p) {
  return load_cvt16_2(p, p + 8);
}
// load 2 x 8 contiguous bf16 (16B each) into a fragment
__device__ __forceinline__ v16bf load_bf8x2(const bf16_t* p0, const bf16_t* p1) {
  v8bf a = *(const v8bf*)p0;
  v8bf b = *(const v8bf*)p1;
  v16bf r;
  #pragma unroll
  for (int i = 0; i < 8; i++) { r[i] = a[i]; r[i + 8] = b[i]; }
  return r;
}

// ---------------------------------------------------------------------------
// Kernel 1: fused QKV projections. z=0: Q=Xto*Wq^T+bq -> Qb[bh][t][64]
//           z=1: K=Xfrom*Wv^T+bv -> Kb[bh][f][64]   (reference swaps names)
//           z=2: V=Xfrom*Wk^T+bk -> Vt[bh][64][f]   (stored transposed)
// grid (32, 16, 3), block 256 (8 waves); wave tile = 16(M) x 64(N), K=1024.
// ---------------------------------------------------------------------------
__global__ __launch_bounds__(256) void qkv_kernel(
    const float* __restrict__ Xto, const float* __restrict__ Xfrom,
    const float* __restrict__ Wq, const float* __restrict__ bq,
    const float* __restrict__ Wv, const float* __restrict__ bv,
    const float* __restrict__ Wk, const float* __restrict__ bk,
    bf16_t* __restrict__ Qb, bf16_t* __restrict__ Kb, bf16_t* __restrict__ Vt) {
  const int lane = threadIdx.x & 31;
  const int wave = threadIdx.x >> 5;
  const int lh   = lane & 15;
  const int lhi  = lane >> 4;
  const int z    = blockIdx.z;

  const float* A; const float* W; const float* bias;
  if (z == 0)      { A = Xto;   W = Wq; bias = bq; }
  else if (z == 1) { A = Xfrom; W = Wv; bias = bv; }
  else             { A = Xfrom; W = Wk; bias = bk; }

  const int m0 = blockIdx.x * 128 + wave * 16;   // row in [0,4096)
  const int n0 = blockIdx.y * 64;                // out feature block (one head)
  const float* rowA = A + (size_t)(m0 + lh) * 1024;
  const float* rowW = W + (size_t)(n0 + lh) * 1024 + lhi * 16;

  v8f acc[4] = {};
  for (int k0 = 0; k0 < 1024; k0 += 32) {
    v16bf a = load_cvt16_2(rowA + k0 + lhi * 16, rowA + k0 + lhi * 16 + 8);
    #pragma unroll
    for (int nt = 0; nt < 4; nt++) {
      v16bf bfrag = load_cvt16(rowW + (size_t)nt * 16 * 1024 + k0);
      acc[nt] = wmma_bf16(a, bfrag, acc[nt]);
    }
  }

  const int h = blockIdx.y;
  #pragma unroll
  for (int nt = 0; nt < 4; nt++) {
    const int n = n0 + nt * 16 + lh;
    const float bb = bias[n];
    const int d = n & 63;
    #pragma unroll
    for (int r = 0; r < 8; r++) {
      const int m = m0 + r + 8 * lhi;
      const int b = m >> 10;
      const int t = m & 1023;
      const int bh = b * 16 + h;
      const float v = acc[nt][r] + bb;
      if (z == 2) {
        Vt[((size_t)bh * 64 + d) * 1024 + t] = (bf16_t)v;
      } else {
        bf16_t* dst = (z == 0) ? Qb : Kb;
        dst[((size_t)bh * 1024 + t) * 64 + d] = (bf16_t)v;
      }
    }
  }
}

// ---------------------------------------------------------------------------
// Kernel 2: masked flash attention per (b,h). Each wave: 16 queries, full
// f=1024 swept in 32-wide steps with online softmax. grid (8, 64), block 256.
// ---------------------------------------------------------------------------
__global__ __launch_bounds__(256) void attn_kernel(
    const bf16_t* __restrict__ Qb, const bf16_t* __restrict__ Kb,
    const bf16_t* __restrict__ Vt, const int* __restrict__ mask,
    bf16_t* __restrict__ ctx) {
  __shared__ bf16_t Plds[8 * 16 * 32];  // per-wave 16x32 bf16 P tile
  const int lane = threadIdx.x & 31;
  const int wave = threadIdx.x >> 5;
  const int lh   = lane & 15;
  const int lhi  = lane >> 4;
  const int bh = blockIdx.y;
  const int b = bh >> 4, h = bh & 15;
  const int t0 = blockIdx.x * 128 + wave * 16;

  // Q fragments (reused across all f tiles): K-dim = head dim (64) = 2 steps
  const bf16_t* Qrow = Qb + ((size_t)bh * 1024 + t0 + lh) * 64;
  v16bf qa0 = load_bf8x2(Qrow + lhi * 16,      Qrow + lhi * 16 + 8);
  v16bf qa1 = load_bf8x2(Qrow + 32 + lhi * 16, Qrow + 40 + lhi * 16);

  const int* mbase = mask + ((size_t)b * 1024 + t0 + 8 * lhi) * 1024;

  v8f o0 = {}, o1 = {}, o2 = {}, o3 = {};
  float mmax[8], ssum[8];
  #pragma unroll
  for (int r = 0; r < 8; r++) { mmax[r] = -1e30f; ssum[r] = 0.0f; }

  bf16_t* Pw = Plds + wave * (16 * 32);
  const bf16_t* Kbase = Kb + (size_t)bh * 1024 * 64;
  const bf16_t* Vbase = Vt + (size_t)bh * 64 * 1024;

  for (int f0 = 0; f0 < 1024; f0 += 32) {
    // scores: two 16x16 tiles, each over K=64 (two 32-deep WMMAs)
    v16bf kb00 = *(const v16bf*)(Kbase + (size_t)(f0 + lh)      * 64      + lhi * 16);
    v16bf kb01 = *(const v16bf*)(Kbase + (size_t)(f0 + lh)      * 64 + 32 + lhi * 16);
    v16bf kb10 = *(const v16bf*)(Kbase + (size_t)(f0 + 16 + lh) * 64      + lhi * 16);
    v16bf kb11 = *(const v16bf*)(Kbase + (size_t)(f0 + 16 + lh) * 64 + 32 + lhi * 16);
    v8f zf = {};
    v8f c0 = wmma_bf16(qa0, kb00, zf); c0 = wmma_bf16(qa1, kb01, c0);
    v8f c1 = wmma_bf16(qa0, kb10, zf); c1 = wmma_bf16(qa1, kb11, c1);

    #pragma unroll
    for (int r = 0; r < 8; r++) {
      const int mk0 = mbase[(size_t)r * 1024 + f0 + lh];
      const int mk1 = mbase[(size_t)r * 1024 + f0 + 16 + lh];
      float v0 = mk0 ? c0[r] * 0.125f : -10000.0f;   // 1/sqrt(64)
      float v1 = mk1 ? c1[r] * 0.125f : -10000.0f;
      float tmax = fmaxf(v0, v1);
      tmax = fmaxf(tmax, __shfl_xor(tmax, 1, 32));
      tmax = fmaxf(tmax, __shfl_xor(tmax, 2, 32));
      tmax = fmaxf(tmax, __shfl_xor(tmax, 4, 32));
      tmax = fmaxf(tmax, __shfl_xor(tmax, 8, 32));   // row max within 16-lane half
      float newm = fmaxf(mmax[r], tmax);
      float fac  = __expf(mmax[r] - newm);
      float p0   = __expf(v0 - newm);
      float p1   = __expf(v1 - newm);
      ssum[r] = ssum[r] * fac + p0 + p1;             // lane-local partial row sum
      mmax[r] = newm;
      o0[r] *= fac; o1[r] *= fac; o2[r] *= fac; o3[r] *= fac;
      const int q = r + 8 * lhi;
      Pw[q * 32 + lh]      = (bf16_t)p0;             // P tile, [q][f_local]
      Pw[q * 32 + 16 + lh] = (bf16_t)p1;
    }
    asm volatile("" ::: "memory");  // same-wave LDS is in-order; fence compiler

    // P (16x32) as A-fragment from LDS, V^T rows as B-fragments
    v16bf pa = load_bf8x2(Pw + lh * 32 + lhi * 16, Pw + lh * 32 + lhi * 16 + 8);
    const bf16_t* vb0 = Vbase + (size_t)lh * 1024 + f0 + lhi * 16;
    o0 = wmma_bf16(pa, *(const v16bf*)(vb0),             o0);
    o1 = wmma_bf16(pa, *(const v16bf*)(vb0 + 16 * 1024), o1);
    o2 = wmma_bf16(pa, *(const v16bf*)(vb0 + 32 * 1024), o2);
    o3 = wmma_bf16(pa, *(const v16bf*)(vb0 + 48 * 1024), o3);
  }

  float inv[8];
  #pragma unroll
  for (int r = 0; r < 8; r++) {
    float s = ssum[r];
    s += __shfl_xor(s, 1, 32);
    s += __shfl_xor(s, 2, 32);
    s += __shfl_xor(s, 4, 32);
    s += __shfl_xor(s, 8, 32);
    inv[r] = 1.0f / s;
  }
  bf16_t* crow = ctx + ((size_t)b * 1024 + t0) * 1024 + h * 64;
  #pragma unroll
  for (int r = 0; r < 8; r++) {
    const int q = r + 8 * lhi;
    bf16_t* cp = crow + (size_t)q * 1024;
    cp[0 * 16 + lh] = (bf16_t)(o0[r] * inv[r]);
    cp[1 * 16 + lh] = (bf16_t)(o1[r] * inv[r]);
    cp[2 * 16 + lh] = (bf16_t)(o2[r] * inv[r]);
    cp[3 * 16 + lh] = (bf16_t)(o3[r] * inv[r]);
  }
}

// ---------------------------------------------------------------------------
// Kernel 3: out = ctx(bf16) @ Wo^T + bo, f32 output. grid (32,16), block 256.
// ---------------------------------------------------------------------------
__global__ __launch_bounds__(256) void oproj_kernel(
    const bf16_t* __restrict__ ctx, const float* __restrict__ Wo,
    const float* __restrict__ bo, float* __restrict__ out) {
  const int lane = threadIdx.x & 31;
  const int wave = threadIdx.x >> 5;
  const int lh   = lane & 15;
  const int lhi  = lane >> 4;
  const int m0 = blockIdx.x * 128 + wave * 16;
  const int n0 = blockIdx.y * 64;
  const bf16_t* rowA = ctx + (size_t)(m0 + lh) * 1024;
  const float*  rowW = Wo  + (size_t)(n0 + lh) * 1024 + lhi * 16;

  v8f acc[4] = {};
  for (int k0 = 0; k0 < 1024; k0 += 32) {
    v16bf a = load_bf8x2(rowA + k0 + lhi * 16, rowA + k0 + lhi * 16 + 8);
    #pragma unroll
    for (int nt = 0; nt < 4; nt++) {
      v16bf bfrag = load_cvt16(rowW + (size_t)nt * 16 * 1024 + k0);
      acc[nt] = wmma_bf16(a, bfrag, acc[nt]);
    }
  }
  #pragma unroll
  for (int nt = 0; nt < 4; nt++) {
    const int n = n0 + nt * 16 + lh;
    const float bb = bo[n];
    #pragma unroll
    for (int r = 0; r < 8; r++) {
      const int m = m0 + r + 8 * lhi;
      out[(size_t)m * 1024 + n] = acc[nt][r] + bb;
    }
  }
}

extern "C" void kernel_launch(void* const* d_in, const int* in_sizes, int n_in,
                              void* d_out, int out_size, void* d_ws, size_t ws_size,
                              hipStream_t stream) {
  const float* Xto   = (const float*)d_in[0];
  const float* Xfrom = (const float*)d_in[1];
  const int*   mask  = (const int*)  d_in[2];
  const float* Wq = (const float*)d_in[3];  const float* bq = (const float*)d_in[4];
  const float* Wk = (const float*)d_in[5];  const float* bk = (const float*)d_in[6];
  const float* Wv = (const float*)d_in[7];  const float* bv = (const float*)d_in[8];
  const float* Wo = (const float*)d_in[9];  const float* bo = (const float*)d_in[10];

  char* ws = (char*)d_ws;
  bf16_t* Qb  = (bf16_t*)(ws);                      // [64][1024][64] bf16, 8 MB
  bf16_t* Kb  = (bf16_t*)(ws + (size_t)8  * 1024 * 1024);
  bf16_t* Vt  = (bf16_t*)(ws + (size_t)16 * 1024 * 1024);  // [64][64][1024]
  bf16_t* ctx = (bf16_t*)(ws + (size_t)24 * 1024 * 1024);  // [4096][1024]
  float* out = (float*)d_out;

  qkv_kernel<<<dim3(32, 16, 3), 256, 0, stream>>>(Xto, Xfrom, Wq, bq, Wv, bv,
                                                  Wk, bk, Qb, Kb, Vt);
  attn_kernel<<<dim3(8, 64), 256, 0, stream>>>(Qb, Kb, Vt, mask, ctx);
  oproj_kernel<<<dim3(32, 16), 256, 0, stream>>>(ctx, Wo, bo, out);
}
